// Model_39676907881305
// MI455X (gfx1250) — matrix operand order repaired and checked by reading, not compiled, not run on hardware
//
// MI455X (gfx1250) flash-attention: out = softmax(Q Kt / 32) V
// Q,K,V: [8192, 1024] f32.  Compute-bound (AI ~2000 flop/byte) -> f16 WMMA
// (v_wmma_f32_16x16x32_f16) with f32 accumulation + online softmax.
// Preprocess: Qh = f16(Q/32), Kh = f16(K), Vt = f16(V^T) in d_ws (48 MB).
// Main kernel: 128 WGs x 512 thr (16 waves): 4 q-subtiles x 4 d-slices,
// Bk=32 key block, K tile double-buffered via async global->LDS,
// B-fragment loads software-pipelined (depth 2) against the WMMAs.

#include <hip/hip_runtime.h>
#include <cstdint>

typedef __attribute__((ext_vector_type(16))) _Float16 v16h;
typedef __attribute__((ext_vector_type(4)))  _Float16 v4h;
typedef __attribute__((ext_vector_type(8)))  float    v8f;
typedef int v4i __attribute__((vector_size(16)));   // b128 payload type

union F16F  { v16h v; uint4 q[2]; };
union F32x8 { v8f  v; uint4 q[2]; };

#define S_TOTAL 8192
#define D_FEAT  1024
#define BK      32
#define QT      4            // q-subtiles of 16 rows per workgroup
#define DSL     4            // d-slices per workgroup
#define DCOLS   256          // columns per d-slice
#define BLOCK   512          // 16 wave32
#define NKB     (S_TOTAL / BK)   // 256
#define KSTRIDE 1032         // halves per K-tile row (1024 + 8 pad, %8==0)
#define VSTRIDE 40           // halves per Vt-tile row (32 + 8 pad, %8==0)
#define PSTRIDE 40           // halves per P-scratch row
#define SLANE   20           // floats per lane in S scratch (16 + 4 pad)

#define KH_OFF   0
#define KH_BYTES (2 * 32 * KSTRIDE * 2)            // 132096 (double buffered)
#define VT_OFF   (KH_OFF + KH_BYTES)
#define VT_BYTES (D_FEAT * VSTRIDE * 2)            // 81920
#define SS_OFF   (VT_OFF + VT_BYTES)
#define SS_BYTES (QT * DSL * 32 * SLANE * 4)       // 40960
#define PS_OFF   (SS_OFF + SS_BYTES)
#define PS_BYTES (16 * 16 * PSTRIDE * 2)           // 20480
#define SMEM_TOTAL (PS_OFF + PS_BYTES)             // 275456 < 320 KB

#define LOG2E 1.4426950408889634f

// ---- CDNA5 async global->LDS path (guarded; sync fallback) ----
#if defined(__HIP_DEVICE_COMPILE__) && defined(__gfx1250__) && defined(__has_builtin)
#if __has_builtin(__builtin_amdgcn_global_load_async_to_lds_b128) && \
    __has_builtin(__builtin_amdgcn_s_wait_asynccnt)
#define HAVE_ASYNC 1
#endif
#endif
#ifndef HAVE_ASYNC
#define HAVE_ASYNC 0
#endif

#if HAVE_ASYNC
#define WAITA(n) __builtin_amdgcn_s_wait_asynccnt(n)
#else
#define WAITA(n) ((void)0)
#endif

__device__ __forceinline__ void cp16(_Float16* lds, const _Float16* g) {
#if HAVE_ASYNC
  // (global v4i* src, shared v4i* dst, imm offset, imm cpol)
  __builtin_amdgcn_global_load_async_to_lds_b128((v4i*)g, (v4i*)lds, 0, 0);
#else
  *(uint4*)lds = *(const uint4*)g;
#endif
}

// ---------------- preprocessing kernels ----------------
__global__ void cvt_scaled(const float4* __restrict__ in, v4h* __restrict__ out,
                           float s, int n4) {
  int i = blockIdx.x * blockDim.x + threadIdx.x;
  if (i < n4) {
    float4 f = in[i];
    v4h h;
    h.x = (_Float16)(f.x * s); h.y = (_Float16)(f.y * s);
    h.z = (_Float16)(f.z * s); h.w = (_Float16)(f.w * s);
    out[i] = h;
  }
}

__global__ void transpose_v(const float* __restrict__ V, _Float16* __restrict__ Vt) {
  __shared__ float tile[32][33];
  const int tx = threadIdx.x, ty = threadIdx.y;
  const int c0 = blockIdx.x * 32, r0 = blockIdx.y * 32;
#pragma unroll
  for (int j = 0; j < 4; ++j)
    tile[ty + j * 8][tx] = V[(size_t)(r0 + ty + j * 8) * D_FEAT + c0 + tx];
  __syncthreads();
#pragma unroll
  for (int j = 0; j < 4; ++j)
    Vt[(size_t)(c0 + ty + j * 8) * S_TOTAL + r0 + tx] = (_Float16)tile[tx][ty + j * 8];
}

// ---------------- main attention kernel ----------------
__global__ __launch_bounds__(BLOCK) void attn_fa(
    const _Float16* __restrict__ Qh,   // [8192][1024], pre-scaled by 1/32
    const _Float16* __restrict__ Kh,   // [8192][1024]
    const _Float16* __restrict__ Vt,   // [1024][8192]  (V transposed)
    float* __restrict__ Out) {
  extern __shared__ char smem[];
  _Float16* khs  = (_Float16*)(smem + KH_OFF);  // [2][32][KSTRIDE]
  _Float16* vts  = (_Float16*)(smem + VT_OFF);  // [1024][VSTRIDE]
  float*    sscr = (float*)   (smem + SS_OFF);  // [QT][DSL][32][SLANE]
  _Float16* pscr = (_Float16*)(smem + PS_OFF);  // [16 waves][16][PSTRIDE]

  const int tid    = threadIdx.x;
  const int wave   = tid >> 5;
  const int lane   = tid & 31;
  const int lane16 = lane & 15;
  const int hi     = lane >> 4;
  const int qi     = wave >> 2;   // 0..3
  const int ds     = wave & 3;    // 0..3
  const int q0     = blockIdx.x * (QT * 16);

  // ---- Q A-fragments (16x32 f16 each), 8 per wave, from global f16 ----
  F16F qA[8];
  {
    const int row = q0 + qi * 16 + lane16;
    const _Float16* qb = Qh + (size_t)row * D_FEAT + ds * DCOLS;
#pragma unroll
    for (int kf = 0; kf < 8; ++kf) {
      const _Float16* p = qb + kf * 32 + hi * 8;
      qA[kf].q[0] = *(const uint4*)(p);
      qA[kf].q[1] = *(const uint4*)(p + 16);
    }
  }

  v8f o[16];
#pragma unroll
  for (int t = 0; t < 16; ++t) o[t] = (v8f)0.0f;
  float mrun[8], lrun[8];
#pragma unroll
  for (int i = 0; i < 8; ++i) { mrun[i] = -1e30f; lrun[i] = 0.0f; }

  _Float16* myp = pscr + wave * 16 * PSTRIDE;

  auto issueK = [&](int kb2, int buf) {
    _Float16* dst = khs + buf * (32 * KSTRIDE);
    const _Float16* src = Kh + (size_t)kb2 * BK * D_FEAT;
#pragma unroll
    for (int j = 0; j < 8; ++j) {
      int c = j * BLOCK + tid;          // 4096 16B chunks
      int row = c >> 7;                 // 0..31
      int col = (c & 127) * 8;          // halves
      cp16(dst + row * KSTRIDE + col, src + row * D_FEAT + col);
    }
  };
  auto issueV = [&](int kb2) {
    const _Float16* src = Vt + (size_t)kb2 * BK;
#pragma unroll
    for (int j = 0; j < 8; ++j) {
      int c = j * BLOCK + tid;          // 4096 16B chunks
      int row = c >> 2;                 // 0..1023 (V columns)
      int cc  = (c & 3) * 8;            // halves within 32-key row
      cp16(vts + row * VSTRIDE + cc, src + (size_t)row * S_TOTAL + cc);
    }
  };

  issueK(0, 0);
  for (int kb = 0; kb < NKB; ++kb) {
    const int cur = kb & 1;
    issueV(kb);
    if (kb + 1 < NKB) { issueK(kb + 1, cur ^ 1); WAITA(16); } else { WAITA(8); }
    __syncthreads();                       // B1: K tile[cur] visible

    // ---- partial S = Qslice * Kslice^T : 2 C-tiles (16x32) ----
    // B-fragment loads pipelined one step ahead of the WMMAs.
    v8f s0 = (v8f)0.0f, s1 = (v8f)0.0f;
    {
      const _Float16* kt0 = khs + cur * (32 * KSTRIDE) +
                            (size_t)lane16 * KSTRIDE + ds * DCOLS + hi * 8;
      const _Float16* kt1 = kt0 + (size_t)16 * KSTRIDE;
      F16F b0[2], b1[2];
      b0[0].q[0] = *(const uint4*)kt0;
      b0[0].q[1] = *(const uint4*)(kt0 + 16);
      b1[0].q[0] = *(const uint4*)kt1;
      b1[0].q[1] = *(const uint4*)(kt1 + 16);
#pragma unroll
      for (int kf = 0; kf < 8; ++kf) {
        if (kf + 1 < 8) {
          const _Float16* p0 = kt0 + (kf + 1) * 32;
          const _Float16* p1 = kt1 + (kf + 1) * 32;
          b0[(kf + 1) & 1].q[0] = *(const uint4*)p0;
          b0[(kf + 1) & 1].q[1] = *(const uint4*)(p0 + 16);
          b1[(kf + 1) & 1].q[0] = *(const uint4*)p1;
          b1[(kf + 1) & 1].q[1] = *(const uint4*)(p1 + 16);
        }
        s0 = __builtin_amdgcn_wmma_f32_16x16x32_f16(false, qA[kf].v, false,
                                                    b0[kf & 1].v, (short)0, s0,
                                                    false, false);
        s1 = __builtin_amdgcn_wmma_f32_16x16x32_f16(false, qA[kf].v, false,
                                                    b1[kf & 1].v, (short)0, s1,
                                                    false, false);
      }
    }

    {   // store partial S (80B/lane, padded)
      float* dst = sscr + ((qi * DSL + ds) * 32 + lane) * SLANE;
      F32x8 u0, u1; u0.v = s0; u1.v = s1;
      ((uint4*)dst)[0] = u0.q[0]; ((uint4*)dst)[1] = u0.q[1];
      ((uint4*)dst)[2] = u1.q[0]; ((uint4*)dst)[3] = u1.q[1];
    }
    __syncthreads();                       // B2: partials ready

    // ---- reduce partials across the 4 d-slices ----
    v8f f0 = (v8f)0.0f, f1 = (v8f)0.0f;
#pragma unroll
    for (int dd = 0; dd < DSL; ++dd) {
      const float* src = sscr + ((qi * DSL + dd) * 32 + lane) * SLANE;
      F32x8 u0, u1;
      u0.q[0] = ((const uint4*)src)[0]; u0.q[1] = ((const uint4*)src)[1];
      u1.q[0] = ((const uint4*)src)[2]; u1.q[1] = ((const uint4*)src)[3];
      f0 += u0.v; f1 += u1.v;
    }

    // ---- online softmax (rows live in (vgpr i, lane-half)) ----
#pragma unroll
    for (int i = 0; i < 8; ++i) {
      float a = f0[i], b = f1[i];
      float mx = fmaxf(a, b);
      mx = fmaxf(mx, __shfl_xor(mx, 1));
      mx = fmaxf(mx, __shfl_xor(mx, 2));
      mx = fmaxf(mx, __shfl_xor(mx, 4));
      mx = fmaxf(mx, __shfl_xor(mx, 8));
      float mnew  = fmaxf(mrun[i], mx);
      float alpha = __builtin_amdgcn_exp2f((mrun[i] - mnew) * LOG2E);
      mrun[i] = mnew;
      float p0 = __builtin_amdgcn_exp2f((a - mnew) * LOG2E);
      float p1 = __builtin_amdgcn_exp2f((b - mnew) * LOG2E);
      float rs = p0 + p1;
      rs += __shfl_xor(rs, 1);
      rs += __shfl_xor(rs, 2);
      rs += __shfl_xor(rs, 4);
      rs += __shfl_xor(rs, 8);
      lrun[i] = lrun[i] * alpha + rs;
#pragma unroll
      for (int t = 0; t < 16; ++t) o[t][i] *= alpha;
      const int r = i + hi * 8;
      myp[r * PSTRIDE + lane16]      = (_Float16)p0;
      myp[r * PSTRIDE + 16 + lane16] = (_Float16)p1;
    }

    // ---- P: C-layout -> A-layout via wave-private LDS round trip ----
    F16F pA;
    {
      const _Float16* pp = myp + lane16 * PSTRIDE;
      pA.q[0] = *(const uint4*)(pp + hi * 8);
      pA.q[1] = *(const uint4*)(pp + 16 + hi * 8);
    }

    if (kb + 1 < NKB) { WAITA(8); } else { WAITA(0); }
    __syncthreads();                       // B3: V tile visible

    // ---- O += P * V : 16 N-tiles, B loads pipelined depth 2 ----
    {
      const _Float16* vbase =
          vts + ((size_t)(ds * DCOLS) + lane16) * VSTRIDE + hi * 8;
      F16F bv[2];
      bv[0].q[0] = *(const uint4*)vbase;
      bv[0].q[1] = *(const uint4*)(vbase + 16);
#pragma unroll
      for (int t = 0; t < 16; ++t) {
        if (t + 1 < 16) {
          const _Float16* pv = vbase + (size_t)((t + 1) * 16) * VSTRIDE;
          bv[(t + 1) & 1].q[0] = *(const uint4*)pv;
          bv[(t + 1) & 1].q[1] = *(const uint4*)(pv + 16);
        }
        o[t] = __builtin_amdgcn_wmma_f32_16x16x32_f16(false, pA.v, false,
                                                      bv[t & 1].v, (short)0,
                                                      o[t], false, false);
      }
    }
    __syncthreads();                       // B4: all V reads done before reuse
  }

  // ---- epilogue: O / l, store f32 ----
#pragma unroll
  for (int i = 0; i < 8; ++i) {
    const float inv = 1.0f / lrun[i];
    const int row = q0 + qi * 16 + i + hi * 8;
    float* orow = Out + (size_t)row * D_FEAT + ds * DCOLS + lane16;
#pragma unroll
    for (int t = 0; t < 16; ++t) orow[t * 16] = o[t][i] * inv;
  }
}

// ---------------- host launcher ----------------
extern "C" void kernel_launch(void* const* d_in, const int* in_sizes, int n_in,
                              void* d_out, int out_size, void* d_ws, size_t ws_size,
                              hipStream_t stream) {
  const float* Q = (const float*)d_in[0];
  const float* K = (const float*)d_in[1];
  const float* V = (const float*)d_in[2];
  float* Out = (float*)d_out;

  // ws layout: Qh (16 MB) | Kh (16 MB) | Vt (16 MB)  -- needs 48 MB scratch
  char* ws = (char*)d_ws;
  const size_t HB = (size_t)S_TOTAL * D_FEAT * sizeof(_Float16);
  _Float16* Qh = (_Float16*)(ws);
  _Float16* Kh = (_Float16*)(ws + HB);
  _Float16* Vt = (_Float16*)(ws + 2 * HB);

  const int n4 = S_TOTAL * D_FEAT / 4;
  cvt_scaled<<<n4 / 256, 256, 0, stream>>>((const float4*)Q, (v4h*)Qh,
                                           1.0f / 32.0f, n4);
  cvt_scaled<<<n4 / 256, 256, 0, stream>>>((const float4*)K, (v4h*)Kh,
                                           1.0f, n4);
  transpose_v<<<dim3(D_FEAT / 32, S_TOTAL / 32), dim3(32, 8), 0, stream>>>(V, Vt);

  (void)hipFuncSetAttribute((const void*)attn_fa,
                            hipFuncAttributeMaxDynamicSharedMemorySize,
                            SMEM_TOTAL);
  attn_fa<<<S_TOTAL / (QT * 16), BLOCK, SMEM_TOTAL, stream>>>(Qh, Kh, Vt, Out);
}
